// JustAttentionDropOutGAT_50130858279705
// MI455X (gfx1250) — matrix operand order repaired and checked by reading, not compiled, not run on hardware
//
#include <hip/hip_runtime.h>
#include <hip/hip_bf16.h>

// ---------------------------------------------------------------------------
// Problem constants (from the reference):  B=4, N=256, T=16, H=128, HEADS=4
// M = B*N = 1024, MT = M*T = 16384, FIN=2, 5 transformer layers.
// ---------------------------------------------------------------------------
#define T_DIM   16
#define M_DIM   1024
#define H_DIM   128
#define N_HEADS 4
#define MT      (M_DIM * T_DIM)     // 16384 rows for all big GEMMs

typedef __attribute__((ext_vector_type(16))) _Float16 v16h;
typedef __attribute__((ext_vector_type(8)))  _Float16 f16x8;
typedef __attribute__((ext_vector_type(4)))  _Float16 f16x4;
typedef __attribute__((ext_vector_type(8)))  float    v8f;
typedef __attribute__((ext_vector_type(4)))  float    f32x4;

// ---------------------------------------------------------------------------
// WMMA helpers (CDNA5: D = A(16x32,f16) * B(32x16,f16) + C(16x16,f32))
// ---------------------------------------------------------------------------
__device__ __forceinline__ v8f wmma_f16(v16h a, v16h b, v8f c) {
  return __builtin_amdgcn_wmma_f32_16x16x32_f16(
      /*neg_a=*/false, a, /*neg_b=*/false, b,
      /*c_mod=*/(short)0, c, /*reuse_a=*/false, /*reuse_b=*/false);
}

// A fragment: 16x32 tile, row-major in LDS, row stride lda (halfs, lda%8==0).
// lane m = lane&15; lanes<16 hold K 0..7 & 16..23, lanes>=16 hold 8..15 & 24..31
// Contiguous per lane -> two ds_load_b128.
__device__ __forceinline__ v16h load_frag_a(const _Float16* As, int lda) {
  int lane = threadIdx.x & 31;
  int m  = lane & 15;
  int k0 = (lane < 16) ? 0 : 8;
  f16x8 lo = *(const f16x8*)(As + m * lda + k0);
  f16x8 hi = *(const f16x8*)(As + m * lda + k0 + 16);
  return __builtin_shufflevector(lo, hi, 0, 1, 2, 3, 4, 5, 6, 7,
                                         8, 9, 10, 11, 12, 13, 14, 15);
}

// B fragment from an N-MAJOR (transposed) LDS tile Bt[n][k], stride ldt (halfs,
// ldt%8==0).  lane n = lane&15; lanes<16 hold K 0..7 & 16..23, lanes>=16 hold 8..15 & 24..31.
// Contiguous per lane -> two ds_load_b128.
__device__ __forceinline__ v16h load_frag_bt(const _Float16* Bt, int ldt) {
  int lane = threadIdx.x & 31;
  int n  = lane & 15;
  int k0 = (lane < 16) ? 0 : 8;
  f16x8 lo = *(const f16x8*)(Bt + n * ldt + k0);
  f16x8 hi = *(const f16x8*)(Bt + n * ldt + k0 + 16);
  return __builtin_shufflevector(lo, hi, 0, 1, 2, 3, 4, 5, 6, 7,
                                         8, 9, 10, 11, 12, 13, 14, 15);
}

// ---------------------------------------------------------------------------
// Generic WMMA GEMM:  C[MM,N] = act( A[MM,K] * B[K,N] + bias )
// f32 inputs converted to f16 in LDS (B staged N-major), f32 accumulation.
// Block: 256 thr = 8 waves = 4(M) x 2(N); block tile 128 x 64; K step 32.
// FULLK=true  : K % 32 == 0 -> unguarded float4 staging (b128 loads).
// FULLK=false : guarded scalar staging (only the FIN=2 projection).
// ---------------------------------------------------------------------------
#define GT_M 128
#define GT_N 64
#define LDA  40     // halfs, 80 B row stride (16B-aligned fragment loads)
#define LDBT 40     // halfs, k-stride for N-major B tile

template <bool FULLK>
__global__ __launch_bounds__(256)
void gemm_wmma_kernel(const float* __restrict__ A, const float* __restrict__ B,
                      float* __restrict__ C, int MM, int N, int K,
                      const float* __restrict__ bias, int act) {
  __shared__ __align__(16) _Float16 As[GT_M * LDA];     // 10 KB
  __shared__ __align__(16) _Float16 Bt[GT_N * LDBT];    //  5 KB

  const int tid = threadIdx.x;
  const int wid = tid >> 5;
  const int wm  = wid & 3;          // 0..3 -> 32-row block
  const int wn  = wid >> 2;         // 0..1 -> 32-col block
  const int m0  = blockIdx.y * GT_M;
  const int n0  = blockIdx.x * GT_N;

  v8f zero = {};
  v8f acc[2][2];
  acc[0][0] = zero; acc[0][1] = zero; acc[1][0] = zero; acc[1][1] = zero;

  for (int kk = 0; kk < K; kk += 32) {
    if (FULLK) {
      // A tile 128x32: 1024 float4 loads, packed f16x4 LDS stores
#pragma unroll
      for (int e = 0; e < 4; ++e) {
        int idx = tid + e * 256;            // 0..1023
        int r = idx >> 3, c4 = idx & 7;
        f32x4 a4 = *(const f32x4*)&A[(size_t)(m0 + r) * K + kk + c4 * 4];
        f16x4 h4;
        h4[0] = (_Float16)a4[0]; h4[1] = (_Float16)a4[1];
        h4[2] = (_Float16)a4[2]; h4[3] = (_Float16)a4[3];
        *(f16x4*)&As[r * LDA + c4 * 4] = h4;
      }
      // B tile 32x64 -> transposed (N-major) LDS
#pragma unroll
      for (int e = 0; e < 2; ++e) {
        int idx = tid + e * 256;            // 0..511
        int r = idx >> 4, c4 = idx & 15;    // r = k row, c4*4 = n
        f32x4 b4 = *(const f32x4*)&B[(size_t)(kk + r) * N + n0 + c4 * 4];
#pragma unroll
        for (int j = 0; j < 4; ++j)
          Bt[(c4 * 4 + j) * LDBT + r] = (_Float16)b4[j];
      }
      if (kk + 32 < K)   // hint next A tile into cache (global_prefetch_b8)
        __builtin_prefetch(&A[(size_t)(m0 + (tid >> 1)) * K + kk + 32], 0, 1);
    } else {
      // Guarded scalar staging (K not multiple of 32; zero-pad)
#pragma unroll
      for (int e = 0; e < 16; ++e) {
        int idx = tid + e * 256;
        int r = idx >> 5, c = idx & 31;
        int gk = kk + c;
        As[r * LDA + c] =
            (gk < K) ? (_Float16)A[(size_t)(m0 + r) * K + gk] : (_Float16)0.f;
      }
#pragma unroll
      for (int e = 0; e < 8; ++e) {
        int idx = tid + e * 256;
        int r = idx >> 6, c = idx & 63;     // r = k row, c = n
        int gk = kk + r;
        Bt[c * LDBT + r] =
            (gk < K) ? (_Float16)B[(size_t)gk * N + n0 + c] : (_Float16)0.f;
      }
    }
    __syncthreads();

    v16h a0 = load_frag_a(As + (wm * 32 + 0)  * LDA, LDA);
    v16h a1 = load_frag_a(As + (wm * 32 + 16) * LDA, LDA);
    v16h b0 = load_frag_bt(Bt + (wn * 32 + 0)  * LDBT, LDBT);
    v16h b1 = load_frag_bt(Bt + (wn * 32 + 16) * LDBT, LDBT);
    acc[0][0] = wmma_f16(a0, b0, acc[0][0]);
    acc[0][1] = wmma_f16(a0, b1, acc[0][1]);
    acc[1][0] = wmma_f16(a1, b0, acc[1][0]);
    acc[1][1] = wmma_f16(a1, b1, acc[1][1]);
    __syncthreads();
  }

  const int lane = tid & 31;
  const int col  = lane & 15;
  const int mb   = (lane < 16) ? 0 : 8;
#pragma unroll
  for (int i = 0; i < 2; ++i)
#pragma unroll
    for (int j = 0; j < 2; ++j)
#pragma unroll
      for (int r = 0; r < 8; ++r) {
        int row = m0 + wm * 32 + i * 16 + mb + r;
        int n   = n0 + wn * 32 + j * 16 + col;
        float v = acc[i][j][r];
        if (bias) v += bias[n];
        if (act == 1) v = fmaxf(v, 0.f);
        C[(size_t)row * N + n] = v;
      }
}

// ---------------------------------------------------------------------------
// mask[t][m] from ego_mask[b][t][n]  (m = b*256 + n)
// ---------------------------------------------------------------------------
__global__ void mask_kernel(const unsigned char* __restrict__ ego,
                            float* __restrict__ mask) {
  int g = blockIdx.x * blockDim.x + threadIdx.x;
  if (g >= T_DIM * M_DIM) return;
  int t = g / M_DIM, mm = g % M_DIM;
  int b = mm >> 8, n = mm & 255;
  mask[g] = ego[(b * T_DIM + t) * 256 + n] ? 1.f : 0.f;
}

// ---------------------------------------------------------------------------
// GAT per-node scores: ss/sd[t,m,head] = <h[t,m,head,:], a_src/dst[head,:]>
// ---------------------------------------------------------------------------
__global__ void gat_scores_kernel(const float* __restrict__ h,
                                  const float* __restrict__ asrc,
                                  const float* __restrict__ adst,
                                  float* __restrict__ ss, float* __restrict__ sd) {
  int g = blockIdx.x * blockDim.x + threadIdx.x;   // T*M*HEADS
  if (g >= T_DIM * M_DIM * N_HEADS) return;
  int head = g & 3, tm = g >> 2;
  const float* hp = h + (size_t)tm * (N_HEADS * H_DIM) + head * H_DIM;
  const float* as = asrc + head * H_DIM;
  const float* ad = adst + head * H_DIM;
  float s0 = 0.f, s1 = 0.f;
#pragma unroll 4
  for (int d4 = 0; d4 < H_DIM / 4; ++d4) {
    f32x4 hv = *(const f32x4*)&hp[d4 * 4];
    f32x4 av = *(const f32x4*)&as[d4 * 4];
    f32x4 dv = *(const f32x4*)&ad[d4 * 4];
#pragma unroll
    for (int j = 0; j < 4; ++j) { s0 += hv[j] * av[j]; s1 += hv[j] * dv[j]; }
  }
  int o = tm * N_HEADS + head;
  ss[o] = s0; sd[o] = s1;
}

// ---------------------------------------------------------------------------
// Segment-softmax row stats: online max / sum-exp over incoming edges j.
// E[t,i,j] = (adj[t,j,i]!=0 & m_i & m_j) | (i==j & m_i)
// ---------------------------------------------------------------------------
__global__ void gat_rowstats_kernel(const float* __restrict__ adj,
                                    const float* __restrict__ mask,
                                    const float* __restrict__ ss,
                                    const float* __restrict__ sd,
                                    float* __restrict__ rmax,
                                    float* __restrict__ rsum) {
  int i    = blockIdx.x * blockDim.x + threadIdx.x;   // 0..1023
  int head = blockIdx.y;
  int t    = blockIdx.z;
  float mi  = mask[t * M_DIM + i];
  float sdi = sd[(t * M_DIM + i) * N_HEADS + head];
  const float* adjt = adj + (size_t)t * M_DIM * M_DIM;
  float mx = -3.0e38f, sum = 0.f;
  for (int j = 0; j < M_DIM; ++j) {
    float mj = mask[t * M_DIM + j];
    bool e = ((adjt[(size_t)j * M_DIM + i] != 0.f) && mi > 0.5f && mj > 0.5f) ||
             (i == j && mi > 0.5f);
    if (e) {
      float l = sdi + ss[(t * M_DIM + j) * N_HEADS + head];
      l = (l > 0.f) ? l : 0.2f * l;                 // leaky_relu(0.2)
      if (l > mx) { sum *= __expf(mx - l); mx = l; }
      sum += __expf(l - mx);
    }
  }
  int o = (t * M_DIM + i) * N_HEADS + head;
  rmax[o] = mx; rsum[o] = sum;
}

// ---------------------------------------------------------------------------
// GAT aggregation: out[t,i,:] = relu( mean_head( sum_j alpha*h[t,j,head,:] ) + b ) * m_i
// Block = 128 thr (4 waves = 4 heads), one 16-row i-tile per (t, block).
// P tiles (16x32 f16) materialized on the fly; h tile staged d-major so the
// 8 B-fragments per chunk are contiguous ds_load_b128 pairs.
// ---------------------------------------------------------------------------
__global__ __launch_bounds__(128)
void gat_aggregate_kernel(const float* __restrict__ h,
                          const float* __restrict__ adj,
                          const float* __restrict__ mask,
                          const float* __restrict__ rmax,
                          const float* __restrict__ rsum,
                          const float* __restrict__ ss,
                          const float* __restrict__ sd,
                          const float* __restrict__ bias,
                          float* __restrict__ xout) {
  // Shared: Ph[4][16*32] f16 (4 KB) + Ht[4][128*40] f16 (40 KB); Outs reuses.
  __shared__ __align__(16) float smemf[11264];          // 44 KB
  _Float16* Ph = (_Float16*)smemf;                      // 4 * 512 halfs
  _Float16* Ht = Ph + N_HEADS * 16 * 32;                // 4 * 128*40 halfs
  float*    Outs = smemf;                               // 4 * 2048 floats (reuse)

  const int tid  = threadIdx.x;
  const int lane = tid & 31;
  const int head = tid >> 5;
  const int i0   = blockIdx.x * 16;
  const int t    = blockIdx.y;
  const float* adjt = adj + (size_t)t * M_DIM * M_DIM;

  v8f zero = {};
  v8f acc[8];
#pragma unroll
  for (int nf = 0; nf < 8; ++nf) acc[nf] = zero;

  for (int jc = 0; jc < M_DIM; jc += 32) {
    // Stage h tile for all 4 heads, d-major: Ht[head][d][jj], f32->f16.
    // 4096 float4 reads (coalesced over d), transposed u16 LDS writes.
#pragma unroll 4
    for (int e = 0; e < 32; ++e) {
      int idx = tid + e * 128;              // 0..4095
      int d4 = idx & 31;
      int jj = (idx >> 5) & 31;
      int hh = idx >> 10;
      f32x4 hv = *(const f32x4*)&h[(size_t)(t * M_DIM + jc + jj) * (N_HEADS * H_DIM) +
                                   hh * H_DIM + d4 * 4];
#pragma unroll
      for (int j = 0; j < 4; ++j)
        Ht[hh * (128 * LDBT) + (d4 * 4 + j) * LDBT + jj] = (_Float16)hv[j];
    }
    // P tile for this wave's head: column jj = lane, rows i = 0..15
    {
      int jj = lane, j = jc + jj;
      float mj  = mask[t * M_DIM + j];
      float ssj = ss[(t * M_DIM + j) * N_HEADS + head];
#pragma unroll
      for (int i = 0; i < 16; ++i) {
        int ig = i0 + i;
        float mi = mask[t * M_DIM + ig];
        bool e = ((adjt[(size_t)j * M_DIM + ig] != 0.f) && mi > 0.5f && mj > 0.5f) ||
                 (ig == j && mi > 0.5f);
        float p = 0.f;
        if (e) {
          int o = (t * M_DIM + ig) * N_HEADS + head;
          float l = sd[o] + ssj;
          l = (l > 0.f) ? l : 0.2f * l;
          float s = rsum[o];
          if (s > 0.f) p = __expf(l - rmax[o]) / s;
        }
        Ph[head * (16 * 32) + i * 32 + jj] = (_Float16)p;
      }
    }
    __syncthreads();
    v16h a = load_frag_a(Ph + head * (16 * 32), 32);
#pragma unroll
    for (int nf = 0; nf < 8; ++nf) {
      v16h b = load_frag_bt(Ht + head * (128 * LDBT) + (nf * 16) * LDBT, LDBT);
      acc[nf] = wmma_f16(a, b, acc[nf]);
    }
    __syncthreads();
  }

  // Per-head 16x128 result -> LDS (reuses staging space; barrier above covers)
  {
    int col = lane & 15;
    int mb  = (lane < 16) ? 0 : 8;
#pragma unroll
    for (int nf = 0; nf < 8; ++nf)
#pragma unroll
      for (int r = 0; r < 8; ++r)
        Outs[head * (16 * 128) + (mb + r) * 128 + nf * 16 + col] = acc[nf][r];
  }
  __syncthreads();

  // Head mean + bias + relu + mask -> next layer activations
#pragma unroll
  for (int e = 0; e < 16; ++e) {
    int idx = tid + e * 128;          // 0..2047
    int d = idx & 127, i = idx >> 7;
    int ig = i0 + i;
    float v = 0.25f * (Outs[0 * 2048 + i * 128 + d] + Outs[1 * 2048 + i * 128 + d] +
                       Outs[2 * 2048 + i * 128 + d] + Outs[3 * 2048 + i * 128 + d]) +
              bias[d];
    v = fmaxf(v, 0.f) * mask[t * M_DIM + ig];
    xout[(size_t)(t * M_DIM + ig) * H_DIM + d] = v;
  }
}

// ---------------------------------------------------------------------------
// xseq[m][t][d] = x[t][m][d] + sinusoidal_pe(t, d)
// ---------------------------------------------------------------------------
__global__ void build_xseq_kernel(const float* __restrict__ x,
                                  float* __restrict__ xseq) {
  int g = blockIdx.x * blockDim.x + threadIdx.x;   // M*T*H
  if (g >= MT * H_DIM) return;
  int d = g & 127, t = (g >> 7) & 15, mnode = g >> 11;
  int pair = d >> 1;
  float div = __expf(-(float)(2 * pair) * 0.07195578f);   // ln(1e4)/128
  float ang = (float)t * div;
  float pe  = (d & 1) ? __cosf(ang) : __sinf(ang);
  xseq[g] = x[(size_t)(t * M_DIM + mnode) * H_DIM + d] + pe;
}

// ---------------------------------------------------------------------------
// Transformer attention per (node, head): 16x16 scores via WMMA, softmax in
// LDS, P@V via 2 WMMAs.  Block = 128 thr = 4 waves = 4 heads, one node/block.
// V staged d-major so P@V B-fragments are contiguous ds_load_b128 pairs.
// ---------------------------------------------------------------------------
__global__ __launch_bounds__(128)
void attention_kernel(const float* __restrict__ q, const float* __restrict__ k,
                      const float* __restrict__ v, float* __restrict__ outb) {
  __shared__ __align__(16) _Float16 Qs[N_HEADS][16 * 32];
  __shared__ __align__(16) _Float16 Ks[N_HEADS][16 * 32];
  __shared__ __align__(16) _Float16 Vt[N_HEADS][32 * LDBT];  // [d][t'], t'=16..31 zero
  __shared__ __align__(16) _Float16 Ps[N_HEADS][16 * 32];    // cols 16..31 zero
  __shared__ float Ss[N_HEADS][16 * 16];

  const int tid  = threadIdx.x;
  const int lane = tid & 31;
  const int head = tid >> 5;
  const int m    = blockIdx.x;
  const size_t base = (size_t)m * T_DIM * H_DIM;

#pragma unroll
  for (int e = 0; e < 16; ++e) {
    int idx = lane + e * 32;
    int d = idx & 31, t = idx >> 5;
    Qs[head][t * 32 + d] = (_Float16)q[base + t * H_DIM + head * 32 + d];
    Ks[head][t * 32 + d] = (_Float16)k[base + t * H_DIM + head * 32 + d];
    Vt[head][d * LDBT + t]      = (_Float16)v[base + t * H_DIM + head * 32 + d];
    Vt[head][d * LDBT + t + 16] = (_Float16)0.f;
  }
  __syncthreads();

  v8f zero = {};
  v16h aq = load_frag_a(Qs[head], 32);
  v16h bk = load_frag_a(Ks[head], 32);   // A-layout load of K == fragment of K^T
  v8f  sc = wmma_f16(aq, bk, zero);

  const float scale = 0.17677669529f;    // 1/sqrt(32)
  {
    int col = lane & 15;
    int mb  = (lane < 16) ? 0 : 8;
#pragma unroll
    for (int r = 0; r < 8; ++r) Ss[head][(mb + r) * 16 + col] = sc[r] * scale;
  }
  __syncthreads();

  if (lane < 16) {                       // one softmax row per lane
    float mx = -3.0e38f;
#pragma unroll
    for (int j = 0; j < 16; ++j) mx = fmaxf(mx, Ss[head][lane * 16 + j]);
    float ex[16]; float sum = 0.f;
#pragma unroll
    for (int j = 0; j < 16; ++j) { ex[j] = __expf(Ss[head][lane * 16 + j] - mx); sum += ex[j]; }
    float inv = 1.f / sum;
#pragma unroll
    for (int j = 0; j < 16; ++j) Ps[head][lane * 32 + j] = (_Float16)(ex[j] * inv);
#pragma unroll
    for (int j = 16; j < 32; ++j) Ps[head][lane * 32 + j] = (_Float16)0.f;
  }
  __syncthreads();

  v16h ap = load_frag_a(Ps[head], 32);
#pragma unroll
  for (int nf = 0; nf < 2; ++nf) {
    v16h bv = load_frag_bt(Vt[head] + (nf * 16) * LDBT, LDBT);
    v8f  o  = wmma_f16(ap, bv, zero);
    int col = lane & 15;
    int mb  = (lane < 16) ? 0 : 8;
#pragma unroll
    for (int r = 0; r < 8; ++r)
      outb[base + (mb + r) * H_DIM + head * 32 + nf * 16 + col] = o[r];
  }
}

// ---------------------------------------------------------------------------
// x = LayerNorm(x + r) * s + b   (wave per 128-wide row, float4 accesses,
// shuffle reductions)
// ---------------------------------------------------------------------------
__global__ __launch_bounds__(256)
void ln_residual_kernel(float* __restrict__ x, const float* __restrict__ r,
                        const float* __restrict__ s, const float* __restrict__ b) {
  int wid  = threadIdx.x >> 5;
  int lane = threadIdx.x & 31;
  int row  = blockIdx.x * 8 + wid;
  size_t o = (size_t)row * H_DIM + lane * 4;
  f32x4 xv = *(const f32x4*)&x[o];
  f32x4 rv = *(const f32x4*)&r[o];
  float vals[4], sum = 0.f;
#pragma unroll
  for (int e = 0; e < 4; ++e) { vals[e] = xv[e] + rv[e]; sum += vals[e]; }
#pragma unroll
  for (int off = 16; off > 0; off >>= 1) sum += __shfl_xor(sum, off, 32);
  float mu = sum * (1.f / 128.f);
  float var = 0.f;
#pragma unroll
  for (int e = 0; e < 4; ++e) { float d = vals[e] - mu; var += d * d; }
#pragma unroll
  for (int off = 16; off > 0; off >>= 1) var += __shfl_xor(var, off, 32);
  float rstd = rsqrtf(var * (1.f / 128.f) + 1e-5f);
  f32x4 sv = *(const f32x4*)&s[lane * 4];
  f32x4 bv = *(const f32x4*)&b[lane * 4];
  f32x4 out;
#pragma unroll
  for (int e = 0; e < 4; ++e) out[e] = (vals[e] - mu) * rstd * sv[e] + bv[e];
  *(f32x4*)&x[o] = out;
}

__global__ void copy4_kernel(const float* __restrict__ src, float* __restrict__ dst,
                             int n4) {
  int g = blockIdx.x * blockDim.x + threadIdx.x;
  if (g < n4) *(f32x4*)&dst[g * 4] = *(const f32x4*)&src[g * 4];
}

// ---------------------------------------------------------------------------
// Host orchestration
// ---------------------------------------------------------------------------
static inline void launch_gemm(const float* A, const float* B, float* C,
                               int MM, int N, int K, const float* bias, int act,
                               hipStream_t stream) {
  dim3 grid(N / GT_N, MM / GT_M);
  if ((K & 31) == 0)
    gemm_wmma_kernel<true><<<grid, 256, 0, stream>>>(A, B, C, MM, N, K, bias, act);
  else
    gemm_wmma_kernel<false><<<grid, 256, 0, stream>>>(A, B, C, MM, N, K, bias, act);
}

extern "C" void kernel_launch(void* const* d_in, const int* in_sizes, int n_in,
                              void* d_out, int out_size, void* d_ws, size_t ws_size,
                              hipStream_t stream) {
  const unsigned char* ego   = (const unsigned char*)d_in[0];
  const float* positions     = (const float*)d_in[1];
  const float* adjacency     = (const float*)d_in[2];
  const float* gat1_W        = (const float*)d_in[3];
  const float* gat1_asrc     = (const float*)d_in[4];
  const float* gat1_adst     = (const float*)d_in[5];
  const float* gat1_b        = (const float*)d_in[6];
  const float* gatW          = (const float*)d_in[7];
  const float* gat_asrc      = (const float*)d_in[8];
  const float* gat_adst      = (const float*)d_in[9];
  const float* gat_b         = (const float*)d_in[10];
  const float* Wqkv          = (const float*)d_in[11];
  const float* bqkv          = (const float*)d_in[12];
  const float* Wo            = (const float*)d_in[13];
  const float* bo            = (const float*)d_in[14];
  const float* ln1_s         = (const float*)d_in[15];
  const float* ln1_b         = (const float*)d_in[16];
  const float* ln2_s         = (const float*)d_in[17];
  const float* ln2_b         = (const float*)d_in[18];
  const float* Wff1          = (const float*)d_in[19];
  const float* bff1          = (const float*)d_in[20];
  const float* Wff2          = (const float*)d_in[21];
  const float* bff2          = (const float*)d_in[22];

  // Workspace layout (floats)
  float* w = (float*)d_ws;
  size_t off = 0;
  float* mask = w + off; off += (size_t)T_DIM * M_DIM;          // 16K
  float* x0   = w + off; off += (size_t)MT * H_DIM;             // 2M
  float* x1   = w + off; off += (size_t)MT * H_DIM;             // 2M
  float* hbuf = w + off; off += (size_t)MT * N_HEADS * H_DIM;   // 8M
  float* ss   = w + off; off += (size_t)MT * N_HEADS;
  float* sd   = w + off; off += (size_t)MT * N_HEADS;
  float* rmx  = w + off; off += (size_t)MT * N_HEADS;
  float* rsm  = w + off; off += (size_t)MT * N_HEADS;
  float* xseq = w + off; off += (size_t)MT * H_DIM;
  float* qb   = w + off; off += (size_t)MT * H_DIM;
  float* kb   = w + off; off += (size_t)MT * H_DIM;
  float* vb   = w + off; off += (size_t)MT * H_DIM;
  float* attn = x0;            // reuse after GAT stage
  float* tmp  = x1;            // reuse after GAT stage
  float* ff   = hbuf;          // reuse 8M buffer for FFN hidden

  // ---- masks ----
  mask_kernel<<<(T_DIM * M_DIM + 255) / 256, 256, 0, stream>>>(ego, mask);

  // ---- GAT stack (layer 0: FIN=2; layers 1..5: 128) ----
  float* xc = x0;
  float* xn = x1;
  for (int layer = 0; layer < 6; ++layer) {
    const float* A    = (layer == 0) ? positions : xc;
    int K             = (layer == 0) ? 2 : H_DIM;
    const float* W    = (layer == 0) ? gat1_W    : gatW     + (size_t)(layer - 1) * H_DIM * N_HEADS * H_DIM;
    const float* asrc = (layer == 0) ? gat1_asrc : gat_asrc + (size_t)(layer - 1) * N_HEADS * H_DIM;
    const float* adst = (layer == 0) ? gat1_adst : gat_adst + (size_t)(layer - 1) * N_HEADS * H_DIM;
    const float* gb   = (layer == 0) ? gat1_b    : gat_b    + (size_t)(layer - 1) * H_DIM;

    launch_gemm(A, W, hbuf, MT, N_HEADS * H_DIM, K, nullptr, 0, stream);
    gat_scores_kernel<<<(MT * N_HEADS + 255) / 256, 256, 0, stream>>>(hbuf, asrc, adst, ss, sd);
    gat_rowstats_kernel<<<dim3(M_DIM / 256, N_HEADS, T_DIM), 256, 0, stream>>>(
        adjacency, mask, ss, sd, rmx, rsm);
    gat_aggregate_kernel<<<dim3(M_DIM / 16, T_DIM), 128, 0, stream>>>(
        hbuf, adjacency, mask, rmx, rsm, ss, sd, gb, xn);
    float* t2 = xc; xc = xn; xn = t2;
  }

  // ---- positional encoding + [T,M,H] -> [M,T,H] ----
  build_xseq_kernel<<<(MT * H_DIM + 255) / 256, 256, 0, stream>>>(xc, xseq);

  // ---- transformer stack ----
  for (int l = 0; l < 5; ++l) {
    const float* Wq = Wqkv + (size_t)(l * 3 + 0) * H_DIM * H_DIM;
    const float* Wk = Wqkv + (size_t)(l * 3 + 1) * H_DIM * H_DIM;
    const float* Wv = Wqkv + (size_t)(l * 3 + 2) * H_DIM * H_DIM;
    launch_gemm(xseq, Wq, qb, MT, H_DIM, H_DIM, bqkv + (size_t)(l * 3 + 0) * H_DIM, 0, stream);
    launch_gemm(xseq, Wk, kb, MT, H_DIM, H_DIM, bqkv + (size_t)(l * 3 + 1) * H_DIM, 0, stream);
    launch_gemm(xseq, Wv, vb, MT, H_DIM, H_DIM, bqkv + (size_t)(l * 3 + 2) * H_DIM, 0, stream);

    attention_kernel<<<M_DIM, 128, 0, stream>>>(qb, kb, vb, attn);

    launch_gemm(attn, Wo + (size_t)l * H_DIM * H_DIM, tmp, MT, H_DIM, H_DIM,
                bo + (size_t)l * H_DIM, 0, stream);
    ln_residual_kernel<<<MT / 8, 256, 0, stream>>>(xseq, tmp, ln1_s + l * H_DIM, ln1_b + l * H_DIM);

    launch_gemm(xseq, Wff1 + (size_t)l * H_DIM * 4 * H_DIM, ff, MT, 4 * H_DIM, H_DIM,
                bff1 + (size_t)l * 4 * H_DIM, 1 /*relu*/, stream);
    launch_gemm(ff, Wff2 + (size_t)l * 4 * H_DIM * H_DIM, tmp, MT, H_DIM, 4 * H_DIM,
                bff2 + (size_t)l * H_DIM, 0, stream);
    ln_residual_kernel<<<MT / 8, 256, 0, stream>>>(xseq, tmp, ln2_s + l * H_DIM, ln2_b + l * H_DIM);
  }

  // ---- output: [M,T,H] with m=b*N+n is exactly [B,N,T,H] flat ----
  copy4_kernel<<<(MT * H_DIM / 4 + 255) / 256, 256, 0, stream>>>(
      xseq, (float*)d_out, MT * H_DIM / 4);
}
